// SingleGraphModel_16758962389230
// MI455X (gfx1250) — matrix operand order repaired and checked by reading, not compiled, not run on hardware
//
#include <hip/hip_runtime.h>
#include <hip/hip_bf16.h>
#include <math.h>

// ---------------- CDNA5 WMMA types ----------------
typedef __attribute__((ext_vector_type(16))) __bf16 v16bf;
typedef __attribute__((ext_vector_type(8)))  float  v8f;
typedef __attribute__((ext_vector_type(4)))  unsigned u32x4;

#define HIDW 128
#define HEADS 4
#define CH 32
#define NCLS 40
#define NEG_SLOPE 0.2f
#define LDB 136  // bf16 LDS row stride (272B: keeps 16B alignment, spreads banks)

// f32 -> bf16 via native conversion (backend picks v_cvt_*_bf16_f32 if present)
__device__ __forceinline__ unsigned short bf1(float a) {
  union { __bf16 h; unsigned short u; } r;
  r.h = (__bf16)a;
  return r.u;
}
__device__ __forceinline__ unsigned pk_bf16(float a, float b) {
  union { __bf16 h[2]; unsigned u; } r;
  r.h[0] = (__bf16)a;
  r.h[1] = (__bf16)b;
  return r.u;
}

// ordered-uint encoding so unsigned atomicMax == float max
__device__ __forceinline__ unsigned f2ord(float f) {
  unsigned u = __float_as_uint(f);
  return (u & 0x80000000u) ? ~u : (u | 0x80000000u);
}
__device__ __forceinline__ float ord2f(unsigned u) {
  return (u & 0x80000000u) ? __uint_as_float(u & 0x7FFFFFFFu)
                           : __uint_as_float(~u);
}

__device__ __forceinline__ v8f wmma_bf16(v16bf a, v16bf b, v8f c) {
  // D = A(16x32 bf16) * B(32x16 bf16) + C(16x16 f32)
  return __builtin_amdgcn_wmma_f32_16x16x32_bf16(
      false, a, false, b, (short)0, c, false, false);
}

// A-fragment straight from a bf16 LDS tile: two 16B ds_load_b128, no VALU.
// ISA 7.12.2 (16-bit A 16x32): lanes 0-15 hold M=lane; VGPR j: K=half*8+2j,+1;
// VGPR 4+j: K=16+half*8+2j,+1.  Memory order of v16bf matches exactly.
__device__ __forceinline__ v16bf a_frag_lds(const unsigned short* base, int kc) {
  int lane = threadIdx.x & 31;
  int m = lane & 15;
  int half = lane >> 4;
  const unsigned short* p = base + m * LDB + kc * 32;
  union { v16bf v; u32x4 q[2]; } fr;
  fr.q[0] = *(const u32x4*)(p + half * 8);
  fr.q[1] = *(const u32x4*)(p + 16 + half * 8);
  return fr.v;
}

// ---------------- weight packing into B-fragment layout ----------------
// pack[(tn*KC + kc)*32 + lane][j] = bf16( W[kc*32 + (j<8 ? half*8+j : 16+half*8+j-8)][tn*16 + n] )
__global__ void pack_weight_k(const float* __restrict__ W,
                              unsigned short* __restrict__ pack,
                              int K, int NcolsPad, int ldw, int realN) {
  int idx = blockIdx.x * blockDim.x + threadIdx.x;
  int lanes_total = (NcolsPad / 16) * (K / 32) * 32;
  if (idx >= lanes_total) return;
  int lane = idx & 31;
  int kc = (idx >> 5) % (K / 32);
  int tn = (idx >> 5) / (K / 32);
  int n = lane & 15;
  int half = lane >> 4;
  int col = tn * 16 + n;
  unsigned short* dst = pack + (size_t)idx * 16;
#pragma unroll
  for (int j = 0; j < 16; ++j) {
    int k = kc * 32 + (j < 8 ? half * 8 + j : 16 + half * 8 + (j - 8));
    float v = (col < realN) ? W[k * ldw + col] : 0.0f;
    dst[j] = bf1(v);
  }
}

// ---------------- init scratch ----------------
__global__ void init_buffers_k(unsigned* __restrict__ amax,
                               float* __restrict__ denom,
                               float* __restrict__ accum,
                               float* __restrict__ meansum, int N) {
  int idx = blockIdx.x * blockDim.x + threadIdx.x;
  if (idx == 0) meansum[0] = 0.0f;
  if (idx < N * HEADS) { amax[idx] = 0x007FFFFFu; /* ord(-inf) */ denom[idx] = 0.0f; }
  if (idx < N * HIDW) accum[idx] = 0.0f;
}

// ---------------- mean(edge_weight) ----------------
__global__ void __launch_bounds__(256)
mean_reduce_k(const float* __restrict__ ew, float* __restrict__ meansum, int E) {
  __shared__ float sh[256];
  float s = 0.0f;
  for (int i = blockIdx.x * blockDim.x + threadIdx.x; i < E;
       i += gridDim.x * blockDim.x)
    s += ew[i];
  sh[threadIdx.x] = s;
  __syncthreads();
  for (int st = 128; st > 0; st >>= 1) {
    if ((int)threadIdx.x < st) sh[threadIdx.x] += sh[threadIdx.x + st];
    __syncthreads();
  }
  if (threadIdx.x == 0) atomicAdd(meansum, sh[0]);
}

// ---------------- fused: h0 = x*Wpre ; xl = h0*Wl+bl ; xr = h0*Wr+br --------
__global__ void __launch_bounds__(256)
fused_pre_k(const float* __restrict__ x,
            const unsigned short* __restrict__ wpre_p,
            const unsigned short* __restrict__ wl_p,
            const unsigned short* __restrict__ wr_p,
            const float* __restrict__ bl, const float* __restrict__ br,
            float* __restrict__ h0, float* __restrict__ xl,
            float* __restrict__ xr, int N) {
  __shared__ __align__(16) unsigned short shb[16][LDB];  // bf16 A tile
  __shared__ float shf[16][HIDW + 4];                    // f32 h0 tile
  int row0 = blockIdx.x * 16;
  int tid = threadIdx.x;
  int w = tid >> 5, lane = tid & 31;

  // cooperative: x tile -> bf16 LDS (one packed cvt per element, done once)
  for (int idx = tid; idx < 16 * 64; idx += 256) {
    int r = idx >> 6, cp = idx & 63;  // bf16 pair index
    float a = 0.0f, b = 0.0f;
    if (row0 + r < N) {
      const float* src = x + (size_t)(row0 + r) * HIDW + 2 * cp;
      a = src[0]; b = src[1];
    }
    *(unsigned*)&shb[r][2 * cp] = pk_bf16(a, b);
  }
  __syncthreads();

  const v16bf* bpre = (const v16bf*)wpre_p;
  v8f acc = {};
#pragma unroll
  for (int kc = 0; kc < 4; ++kc) {
    v16bf a = a_frag_lds(&shb[0][0], kc);
    acc = wmma_bf16(a, bpre[(w * 4 + kc) * 32 + lane], acc);
  }
  __syncthreads();

  int n = lane & 15, half = lane >> 4;
#pragma unroll
  for (int r = 0; r < 8; ++r) shf[half * 8 + r][w * 16 + n] = acc[r];
  __syncthreads();

  // cooperative: h0 tile -> global (f32, coalesced) + bf16 LDS for stage 2
  for (int idx = tid; idx < 16 * 64; idx += 256) {
    int r = idx >> 6, cp = idx & 63;
    float a = shf[r][2 * cp], b = shf[r][2 * cp + 1];
    if (row0 + r < N) {
      float2 v2; v2.x = a; v2.y = b;
      *(float2*)&h0[(size_t)(row0 + r) * HIDW + 2 * cp] = v2;
    }
    *(unsigned*)&shb[r][2 * cp] = pk_bf16(a, b);
  }
  __syncthreads();

  // stage 2: xl / xr from bf16 h0 tile
  const v16bf* bl_p2 = (const v16bf*)wl_p;
  const v16bf* br_p2 = (const v16bf*)wr_p;
  v8f accl = {}, accr = {};
#pragma unroll
  for (int kc = 0; kc < 4; ++kc) {
    v16bf a = a_frag_lds(&shb[0][0], kc);
    accl = wmma_bf16(a, bl_p2[(w * 4 + kc) * 32 + lane], accl);
    accr = wmma_bf16(a, br_p2[(w * 4 + kc) * 32 + lane], accr);
  }
  int col = w * 16 + n;
  float bvl = bl[col], bvr = br[col];
#pragma unroll
  for (int r = 0; r < 8; ++r) {
    int row = row0 + half * 8 + r;
    if (row < N) {
      xl[(size_t)row * HIDW + col] = accl[r] + bvl;
      xr[(size_t)row * HIDW + col] = accr[r] + bvr;
    }
  }
}

// ---------------- edge pass 1: alpha + segment max ----------------
__global__ void __launch_bounds__(256)
edge_alpha_max_k(const int* __restrict__ ei, const float* __restrict__ ew,
                 const float* __restrict__ xl, const float* __restrict__ xr,
                 const float* __restrict__ We, const float* __restrict__ att,
                 const float* __restrict__ meansum,
                 unsigned* __restrict__ amax, int E, int Ntot) {
  int wave = (int)((blockIdx.x * blockDim.x + threadIdx.x) >> 5);
  int lane = threadIdx.x & 31;
  if (wave >= Ntot) return;
  int src, dst; float ea;
  if (wave < E) { src = ei[wave]; dst = ei[E + wave]; ea = ew[wave]; }
  else { src = dst = wave - E; ea = meansum[0] * (1.0f / (float)E); }
#pragma unroll
  for (int h = 0; h < HEADS; ++h) {
    int i = h * CH + lane;
    float v = xl[(size_t)src * HIDW + i] + xr[(size_t)dst * HIDW + i] + ea * We[i];
    v = (v > 0.0f) ? v : v * NEG_SLOPE;
    float p = att[i] * v;
#pragma unroll
    for (int off = 16; off > 0; off >>= 1) p += __shfl_xor(p, off, 32);
    if (lane == 0) atomicMax(&amax[dst * HEADS + h], f2ord(p));
  }
}

// ---------------- edge pass 2: exp + weighted scatter-sum ----------------
__global__ void __launch_bounds__(256)
edge_accum_k(const int* __restrict__ ei, const float* __restrict__ ew,
             const float* __restrict__ xl, const float* __restrict__ xr,
             const float* __restrict__ We, const float* __restrict__ att,
             const float* __restrict__ meansum,
             const unsigned* __restrict__ amax, float* __restrict__ denom,
             float* __restrict__ accum, int E, int Ntot) {
  int wave = (int)((blockIdx.x * blockDim.x + threadIdx.x) >> 5);
  int lane = threadIdx.x & 31;
  if (wave >= Ntot) return;
  int src, dst; float ea;
  if (wave < E) { src = ei[wave]; dst = ei[E + wave]; ea = ew[wave]; }
  else { src = dst = wave - E; ea = meansum[0] * (1.0f / (float)E); }
#pragma unroll
  for (int h = 0; h < HEADS; ++h) {
    int i = h * CH + lane;
    float xlv = xl[(size_t)src * HIDW + i];
    float v = xlv + xr[(size_t)dst * HIDW + i] + ea * We[i];
    v = (v > 0.0f) ? v : v * NEG_SLOPE;
    float p = att[i] * v;
#pragma unroll
    for (int off = 16; off > 0; off >>= 1) p += __shfl_xor(p, off, 32);
    float am = ord2f(amax[dst * HEADS + h]);
    if (!(am >= -1.0e30f)) am = 0.0f;  // mirror isfinite guard (-inf / NaN -> 0)
    float ex = __expf(p - am);
    atomicAdd(&accum[(size_t)dst * HIDW + i], ex * xlv);
    if (lane == 0) atomicAdd(&denom[dst * HEADS + h], ex);
  }
}

// ---------------- node finalize: softmax-normalize, ELU, residual ----------
__global__ void node_finalize_k(const float* __restrict__ accum,
                                const float* __restrict__ denom,
                                const float* __restrict__ b_gat,
                                const float* __restrict__ h0,
                                float* __restrict__ out, int N) {
  int idx = blockIdx.x * blockDim.x + threadIdx.x;
  if (idx >= N * HIDW) return;
  int n = idx >> 7, i = idx & 127, h = i >> 5;
  float o = accum[idx] / (denom[n * HEADS + h] + 1e-16f) + b_gat[i];
  float e = (o > 0.0f) ? o : (__expf(o) - 1.0f);
  out[idx] = h0[idx] + e;
}

// ---------------- fused MLP: logits = relu(h*W1+b1)*W2 + b2 ----------------
__global__ void __launch_bounds__(256)
mlp_k(const float* __restrict__ h, const unsigned short* __restrict__ w1p,
      const unsigned short* __restrict__ w2p, const float* __restrict__ b1,
      const float* __restrict__ b2, float* __restrict__ logits, int N) {
  __shared__ __align__(16) unsigned short shb[16][LDB];  // bf16 tile (h, then t)
  int row0 = blockIdx.x * 16;
  int tid = threadIdx.x, w = tid >> 5, lane = tid & 31;

  for (int idx = tid; idx < 16 * 64; idx += 256) {
    int r = idx >> 6, cp = idx & 63;
    float a = 0.0f, b = 0.0f;
    if (row0 + r < N) {
      const float* src = h + (size_t)(row0 + r) * HIDW + 2 * cp;
      a = src[0]; b = src[1];
    }
    *(unsigned*)&shb[r][2 * cp] = pk_bf16(a, b);
  }
  __syncthreads();

  const v16bf* bp1 = (const v16bf*)w1p;
  v8f acc = {};
#pragma unroll
  for (int kc = 0; kc < 4; ++kc) {
    v16bf a = a_frag_lds(&shb[0][0], kc);
    acc = wmma_bf16(a, bp1[(w * 4 + kc) * 32 + lane], acc);
  }
  __syncthreads();

  int n = lane & 15, half = lane >> 4;
  float bv = b1[w * 16 + n];
#pragma unroll
  for (int r = 0; r < 8; ++r) {
    float t = acc[r] + bv;
    shb[half * 8 + r][w * 16 + n] = bf1((t > 0.0f) ? t : 0.0f);
  }
  __syncthreads();

  if (w < 3) {  // 3 N-tiles cover 48 cols (40 real)
    const v16bf* bp2 = (const v16bf*)w2p;
    v8f acc2 = {};
#pragma unroll
    for (int kc = 0; kc < 4; ++kc) {
      v16bf a = a_frag_lds(&shb[0][0], kc);
      acc2 = wmma_bf16(a, bp2[(w * 4 + kc) * 32 + lane], acc2);
    }
    int col = w * 16 + n;
    if (col < NCLS) {
      float b2v = b2[col];
#pragma unroll
      for (int r = 0; r < 8; ++r) {
        int row = row0 + half * 8 + r;
        if (row < N) logits[(size_t)row * NCLS + col] = acc2[r] + b2v;
      }
    }
  }
}

// ---------------- host launch ----------------
extern "C" void kernel_launch(void* const* d_in, const int* in_sizes, int n_in,
                              void* d_out, int out_size, void* d_ws,
                              size_t ws_size, hipStream_t stream) {
  const float* x    = (const float*)d_in[0];
  const int*   ei   = (const int*)d_in[1];
  const float* ew   = (const float*)d_in[2];
  const float* Wpre = (const float*)d_in[3];
  const float* Wl   = (const float*)d_in[4];
  const float* bl   = (const float*)d_in[5];
  const float* Wr   = (const float*)d_in[6];
  const float* br   = (const float*)d_in[7];
  const float* We   = (const float*)d_in[8];
  const float* att  = (const float*)d_in[9];
  const float* bg   = (const float*)d_in[10];
  const float* W1   = (const float*)d_in[11];
  const float* b1   = (const float*)d_in[12];
  const float* W2   = (const float*)d_in[13];
  const float* b2   = (const float*)d_in[14];

  const int N = in_sizes[0] / HIDW;
  const int E = in_sizes[2];
  const int Ntot = E + N;

  // workspace carve (256B aligned regions)
  char* ws = (char*)d_ws;
  auto carve = [&](size_t bytes) {
    char* p = ws;
    ws += (bytes + 255) & ~(size_t)255;
    return p;
  };
  unsigned short* wpre_p = (unsigned short*)carve(128 * 128 * 2);
  unsigned short* wl_p   = (unsigned short*)carve(128 * 128 * 2);
  unsigned short* wr_p   = (unsigned short*)carve(128 * 128 * 2);
  unsigned short* w1_p   = (unsigned short*)carve(128 * 128 * 2);
  unsigned short* w2_p   = (unsigned short*)carve(128 * 48 * 2);
  float*    meansum = (float*)carve(sizeof(float));
  float*    h0    = (float*)carve((size_t)N * HIDW * 4);
  float*    xl    = (float*)carve((size_t)N * HIDW * 4);
  float*    xr    = (float*)carve((size_t)N * HIDW * 4);
  unsigned* amax  = (unsigned*)carve((size_t)N * HEADS * 4);
  float*    denom = (float*)carve((size_t)N * HEADS * 4);
  float*    accum = (float*)carve((size_t)N * HIDW * 4);

  float* out_h      = (float*)d_out;                     // [N,128]
  float* out_logits = (float*)d_out + (size_t)N * HIDW;  // [N,40]

  // 1) pack weights into WMMA B-fragment layout (bf16)
  {
    int lanes128 = 8 * 4 * 32;  // (128/16)*(128/32)*32
    int lanes48  = 3 * 4 * 32;
    pack_weight_k<<<(lanes128 + 255) / 256, 256, 0, stream>>>(Wpre, wpre_p, 128, 128, 128, 128);
    pack_weight_k<<<(lanes128 + 255) / 256, 256, 0, stream>>>(Wl,   wl_p,   128, 128, 128, 128);
    pack_weight_k<<<(lanes128 + 255) / 256, 256, 0, stream>>>(Wr,   wr_p,   128, 128, 128, 128);
    pack_weight_k<<<(lanes128 + 255) / 256, 256, 0, stream>>>(W1,   w1_p,   128, 128, 128, 128);
    pack_weight_k<<<(lanes48  + 255) / 256, 256, 0, stream>>>(W2,   w2_p,   128, 48,  NCLS, NCLS);
  }
  // 2) init atomic scratch
  init_buffers_k<<<((size_t)N * HIDW + 255) / 256, 256, 0, stream>>>(amax, denom, accum, meansum, N);
  // 3) mean(edge_weight)
  mean_reduce_k<<<512, 256, 0, stream>>>(ew, meansum, E);
  // 4) fused pre-projection + l/r transforms (WMMA)
  int nblk = (N + 15) / 16;
  fused_pre_k<<<nblk, 256, 0, stream>>>(x, wpre_p, wl_p, wr_p, bl, br, h0, xl, xr, N);
  // 5) edge pass 1: alpha + segment max
  int eblk = (Ntot + 7) / 8;  // one wave32 per edge, 8 waves/block
  edge_alpha_max_k<<<eblk, 256, 0, stream>>>(ei, ew, xl, xr, We, att, meansum, amax, E, Ntot);
  // 6) edge pass 2: exp + weighted scatter
  edge_accum_k<<<eblk, 256, 0, stream>>>(ei, ew, xl, xr, We, att, meansum, amax, denom, accum, E, Ntot);
  // 7) normalize + ELU + residual -> h
  node_finalize_k<<<((size_t)N * HIDW + 255) / 256, 256, 0, stream>>>(accum, denom, bg, h0, out_h, N);
  // 8) fused MLP -> logits (WMMA)
  mlp_k<<<nblk, 256, 0, stream>>>(out_h, w1_p, w2_p, b1, b2, out_logits, N);
}
